// Attention_52390011076956
// MI455X (gfx1250) — compile-verified, matrix operand order checked by reading, not run on hardware
//
#include <hip/hip_runtime.h>

// ---------------------------------------------------------------------------
// MI455X (gfx1250) attention: bf16 WMMA everywhere, flash-style softmax,
// async global->LDS double-buffered staging in both GEMM and flash kernels,
// bit-packed mask (268 MB -> 8.4 MB in-loop traffic).
// Workspace (us16 units): Wv|Wk|Wq|Wo (4*E*E), Xq|Xk|Xv (3*R*E),
//                         Q|K|Vt|O (4*R*E), mask bits (R*S/32 u32). ~123 MB.
// ---------------------------------------------------------------------------

typedef __bf16 v16bf __attribute__((ext_vector_type(16)));
typedef __bf16 v8bf  __attribute__((ext_vector_type(8)));
typedef __bf16 v4bf  __attribute__((ext_vector_type(4)));
typedef float  v8f   __attribute__((ext_vector_type(8)));
typedef unsigned short us16;
typedef unsigned int   u32;

constexpr int   N_B   = 4;
constexpr int   S_LEN = 4096;
constexpr int   E     = 512;
constexpr int   R     = N_B * S_LEN;          // 16384 flattened tokens
constexpr int   MW    = S_LEN / 32;           // mask words per row
constexpr float SM_SCALE = 0.044194173824159216f;  // 1/sqrt(512)

#define DEVI __device__ __forceinline__

DEVI us16 f2bf(float f) {                     // native RTNE convert
  return __builtin_bit_cast(us16, (__bf16)f);
}

// A/B fragment loader for v_wmma_f32_16x16x32_bf16.
// 16-bit A (16x32): lanes 0-15 hold row M=lane, K chunks {0..7,16..23};
// lanes 16-31 hold row M=lane-16, K chunks {8..15,24..31}.  B (32x16) mirrors
// this with "row" = N column.  tile points at element [0][col0]; pitch in us16.
DEVI v16bf load_frag(const us16* tile, int pitch, int col0, int lane) {
  const us16* p = tile + (lane & 15) * pitch + col0 + ((lane >> 4) << 3);
  v8bf c0 = *reinterpret_cast<const v8bf*>(p);        // ds/global b128
  v8bf c1 = *reinterpret_cast<const v8bf*>(p + 16);   // ds/global b128
  return __builtin_shufflevector(c0, c1, 0,1,2,3,4,5,6,7,8,9,10,11,12,13,14,15);
}

// CDNA5 async global->LDS copy, 16B per lane, tracked by ASYNCcnt.
DEVI void async_ld16(u32 lds_off, const us16* gaddr) {
  asm volatile("global_load_async_to_lds_b128 %0, %1, off"
               :: "v"(lds_off), "v"(gaddr) : "memory");
}
DEVI void wait_async0() { asm volatile("s_wait_asynccnt 0x0" ::: "memory"); }

// ---------------------------------------------------------------------------
// Kernel 1: streaming fp32 -> bf16 convert (weights and activations)
// ---------------------------------------------------------------------------
__global__ __launch_bounds__(256) void cvt_kernel(const float* __restrict__ src,
                                                  us16* __restrict__ dst) {
  int i = (blockIdx.x * 256 + threadIdx.x) * 4;
  float4 v = *reinterpret_cast<const float4*>(src + i);
  v4bf pk;
  pk[0] = (__bf16)v.x; pk[1] = (__bf16)v.y;
  pk[2] = (__bf16)v.z; pk[3] = (__bf16)v.w;   // -> v_cvt_pk_bf16_f32 x2
  *reinterpret_cast<v4bf*>(dst + i) = pk;
}

// ---------------------------------------------------------------------------
// Kernel 2: pack mask int32 -> bits via wave32 ballot (32 ints -> 1 word).
// Each wave packs 8 words.
// ---------------------------------------------------------------------------
__global__ __launch_bounds__(256) void mask_pack_kernel(
    const int* __restrict__ mask, u32* __restrict__ Mb) {
  int wave = (blockIdx.x * 256 + threadIdx.x) >> 5;
  int lane = threadIdx.x & 31;
  size_t base = (size_t)wave * 8 * 32;
#pragma unroll
  for (int j = 0; j < 8; ++j) {
    int m = mask[base + j * 32 + lane];
    unsigned long long bm = __ballot(m != 0);   // wave32: low 32 bits
    if (lane == 0) Mb[wave * 8 + j] = (u32)bm;
  }
}

// ---------------------------------------------------------------------------
// Kernel 3: Y[r,e] = sum_d A[r,d] * W[e,d]   (nn.Linear, x @ W^T), all bf16.
// Tile: 128(M) x 64(N), K-step 32, 8 waves.  Both tiles double-buffered in
// LDS and filled with GLOBAL_LOAD_ASYNC_TO_LDS_B128; step k+1 streams while
// step k's 4 WMMAs run.
// out_mode: 0 = bf16 row-major [R][E]
//           1 = bf16 transposed Vt[n][d][s]  (for the P*V B-operand)
//           2 = f32 row-major + bias
// LDS pitch 40 us16 (80 B = 5*16) keeps every b128 access 16B aligned.
// ---------------------------------------------------------------------------
constexpr int KP    = 40;
constexpr int ATILE = 128 * KP;
constexpr int WTILE = 64 * KP;

DEVI void stage_gemm_async(const us16* A, const us16* W,
                           us16* As, us16* Ws, int tid) {
  {
    int row = tid >> 1, c0 = (tid & 1) * 16;
    const us16* src = A + (size_t)row * E + c0;
    u32 dst = (u32)(uintptr_t)(As + row * KP + c0);
    async_ld16(dst, src);
    async_ld16(dst + 16, src + 8);
  }
  {
    int row = tid >> 2, c0 = (tid & 3) * 8;
    async_ld16((u32)(uintptr_t)(Ws + row * KP + c0), W + (size_t)row * E + c0);
  }
}

__global__ __launch_bounds__(256) void gemm512_kernel(
    const us16* __restrict__ A, const us16* __restrict__ Wb,
    const float* __restrict__ bias, void* __restrict__ Yany, int out_mode) {
  __shared__ __align__(16) us16 As[2 * ATILE];
  __shared__ __align__(16) us16 Ws[2 * WTILE];

  const int tid  = threadIdx.x;
  const int lane = tid & 31;
  const int wv   = tid >> 5;                  // 0..7
  const int rbase = blockIdx.x * 128;
  const int nbase = blockIdx.y * 64;
  const int mrow  = wv * 16;
  const int cl = lane & 15, hl = lane >> 4;

  const us16* Abase = A  + (size_t)rbase * E;
  const us16* Wbase = Wb + (size_t)nbase * E;

  v8f acc[4] = {};

  stage_gemm_async(Abase, Wbase, As, Ws, tid);      // prologue: buffer 0

  for (int k0 = 0; k0 < E; k0 += 32) {
    const int b = (k0 >> 5) & 1;
    wait_async0();
    __syncthreads();
    if (k0 + 32 < E)
      stage_gemm_async(Abase + k0 + 32, Wbase + k0 + 32,
                       As + (b ^ 1) * ATILE, Ws + (b ^ 1) * WTILE, tid);

    v16bf af = load_frag(As + b * ATILE + mrow * KP, KP, 0, lane);
#pragma unroll
    for (int nc = 0; nc < 4; ++nc) {
      v16bf bfg = load_frag(Ws + b * WTILE + nc * 16 * KP, KP, 0, lane);
      acc[nc] = __builtin_amdgcn_wmma_f32_16x16x32_bf16(
          false, af, false, bfg, (short)0, acc[nc], false, false);
    }
    // next iteration's wait+barrier protects the buffers
  }

  // ---- epilogue ----
  if (out_mode == 0) {                        // bf16 [R][E]
    us16* Y = (us16*)Yany;
#pragma unroll
    for (int nc = 0; nc < 4; ++nc) {
      int col = nbase + nc * 16 + cl;
#pragma unroll
      for (int g = 0; g < 8; ++g) {
        int row = rbase + mrow + hl * 8 + g;
        Y[(size_t)row * E + col] = f2bf(acc[nc][g]);   // 16 lanes -> 32B contig
      }
    }
  } else if (out_mode == 1) {                 // bf16 Vt[n][d][s]
    us16* Y = (us16*)Yany;
    int n  = rbase / S_LEN;                   // 4096 % 128 == 0: one batch/blk
    int s0 = (rbase + mrow + hl * 8) & (S_LEN - 1);
#pragma unroll
    for (int nc = 0; nc < 4; ++nc) {
      int d = nbase + nc * 16 + cl;
      alignas(16) us16 pk[8];
#pragma unroll
      for (int g = 0; g < 8; ++g) pk[g] = f2bf(acc[nc][g]);
      *reinterpret_cast<uint4*>(Y + ((size_t)n * E + d) * S_LEN + s0) =
          *reinterpret_cast<uint4*>(pk);      // global_store_b128
    }
  } else {                                    // f32 + bias
    float* Y = (float*)Yany;
#pragma unroll
    for (int nc = 0; nc < 4; ++nc) {
      int col = nbase + nc * 16 + cl;
      float b = bias[col];
#pragma unroll
      for (int g = 0; g < 8; ++g) {
        int row = rbase + mrow + hl * 8 + g;
        Y[(size_t)row * E + col] = acc[nc][g] + b;
      }
    }
  }
}

// ---------------------------------------------------------------------------
// Kernel 4: flash attention.  Block = 128 queries (8 waves x 16 rows),
// streams 32 keys/step.  Q fragments resident in VGPRs; O accumulators
// 32 x v8f.  K/V tiles double-buffered in LDS via async-to-LDS; mask read
// as packed bits (8.4 MB total, L2-resident).
// LDS: 2x33 KB (K) + 2x40 KB (Vt) + 10 KB (P) ~ 155 KB.
// ---------------------------------------------------------------------------
constexpr int KSP = 520;   // K tile pitch over d (512+8); 1040 B rows
constexpr int VSP = 40;    // Vt tile pitch over keys (32+8)
constexpr int KTILE = 32 * KSP;
constexpr int VTILE = 512 * VSP;

DEVI void stage_tiles_async(const us16* Kg, const us16* Vg,
                            us16* KsBuf, us16* VsBuf, int tid) {
  {
    int row = tid >> 3, c0 = (tid & 7) * 64;
    const us16* src = Kg + (size_t)row * E + c0;
    u32 dst = (u32)(uintptr_t)(KsBuf + row * KSP + c0);
#pragma unroll
    for (int j = 0; j < 8; ++j) async_ld16(dst + j * 16, src + j * 8);
  }
  {
    int d0 = tid * 2;
#pragma unroll
    for (int rr = 0; rr < 2; ++rr) {
      const us16* src = Vg + (size_t)(d0 + rr) * S_LEN;
      u32 dst = (u32)(uintptr_t)(VsBuf + (d0 + rr) * VSP);
#pragma unroll
      for (int j = 0; j < 4; ++j) async_ld16(dst + j * 16, src + j * 8);
    }
  }
}

__global__ __launch_bounds__(256) void flash_kernel(
    const us16* __restrict__ Qb, const us16* __restrict__ Kb,
    const us16* __restrict__ Vt, const u32* __restrict__ Mb,
    us16* __restrict__ Ob) {
  __shared__ __align__(16) us16 Ks[2 * KTILE];
  __shared__ __align__(16) us16 Vs[2 * VTILE];
  __shared__ __align__(16) us16 Ps[8 * 16 * VSP];

  const int tid  = threadIdx.x;
  const int lane = tid & 31;
  const int wv   = tid >> 5;
  const int n    = blockIdx.y;
  const int qblk = blockIdx.x * 128;
  const int cl = lane & 15, hl = lane >> 4;

  const us16* Kbn = Kb + (size_t)n * S_LEN * E;   // [key][d]
  const us16* Vtn = Vt + (size_t)n * E * S_LEN;   // [d][s]

  // ---- Q fragments resident in registers ----
  v16bf qf[16];
  {
    const us16* qrow = Qb + ((size_t)n * S_LEN + qblk + wv * 16) * E;
#pragma unroll
    for (int kc = 0; kc < 16; ++kc) qf[kc] = load_frag(qrow, E, kc * 32, lane);
  }

  v8f acc[32] = {};
  float m_run[8], l_run[8];
#pragma unroll
  for (int g = 0; g < 8; ++g) { m_run[g] = -3.0e38f; l_run[g] = 0.0f; }

  const u32* Mbase = Mb + ((size_t)n * S_LEN + qblk + wv * 16 + hl * 8) * MW;

  stage_tiles_async(Kbn, Vtn, Ks, Vs, tid);       // prologue: buffer 0

  for (int kv = 0; kv < S_LEN; kv += 32) {
    const int b = (kv >> 5) & 1;
    wait_async0();                 // my async writes for tile kv are done
    __syncthreads();               // everyone's are; prior compute finished

    if (kv + 32 < S_LEN)           // prefetch next tile into other buffer
      stage_tiles_async(Kbn + (size_t)(kv + 32) * E, Vtn + (kv + 32),
                        Ks + (b ^ 1) * KTILE, Vs + (b ^ 1) * VTILE, tid);

    const us16* KsB = Ks + b * KTILE;
    const us16* VsB = Vs + b * VTILE;

    // ---- scores: 16x32 = two 16x16 subtiles over 16 K-steps ----
    v8f s0 = {}, s1 = {};
#pragma unroll
    for (int kc = 0; kc < 16; ++kc) {
      v16bf b0 = load_frag(KsB,            KSP, kc * 32, lane);
      v16bf b1 = load_frag(KsB + 16 * KSP, KSP, kc * 32, lane);
      s0 = __builtin_amdgcn_wmma_f32_16x16x32_bf16(false, qf[kc], false, b0,
                                                   (short)0, s0, false, false);
      s1 = __builtin_amdgcn_wmma_f32_16x16x32_bf16(false, qf[kc], false, b1,
                                                   (short)0, s1, false, false);
    }

    // ---- packed-bit mask + online softmax (row reductions, 16-lane halves) --
    float p0[8], p1[8], alpha[8];
    const int mword = kv >> 5;
#pragma unroll
    for (int g = 0; g < 8; ++g) {
      u32 mw = Mbase[(size_t)g * MW + mword];
      float v0 = ((mw >> cl) & 1u)        ? s0[g] * SM_SCALE : -1.0e30f;
      float v1 = ((mw >> (16 + cl)) & 1u) ? s1[g] * SM_SCALE : -1.0e30f;
      float t = fmaxf(v0, v1);
      t = fmaxf(t, __shfl_xor(t, 1, 32));
      t = fmaxf(t, __shfl_xor(t, 2, 32));
      t = fmaxf(t, __shfl_xor(t, 4, 32));
      t = fmaxf(t, __shfl_xor(t, 8, 32));
      float mnew = fmaxf(m_run[g], t);
      alpha[g]   = __expf(m_run[g] - mnew);
      m_run[g]   = mnew;
      p0[g] = __expf(v0 - mnew);
      p1[g] = __expf(v1 - mnew);
      float rs = p0[g] + p1[g];
      rs += __shfl_xor(rs, 1, 32);
      rs += __shfl_xor(rs, 2, 32);
      rs += __shfl_xor(rs, 4, 32);
      rs += __shfl_xor(rs, 8, 32);
      l_run[g] = l_run[g] * alpha[g] + rs;
    }

    // ---- rescale O accumulators ----
#pragma unroll
    for (int dc = 0; dc < 32; ++dc)
#pragma unroll
      for (int g = 0; g < 8; ++g) acc[dc][g] *= alpha[g];

    // ---- bounce P through per-wave LDS patch (C-layout -> A-layout) ----
    us16* pw = Ps + wv * 16 * VSP;
#pragma unroll
    for (int g = 0; g < 8; ++g) {
      pw[(hl * 8 + g) * VSP + cl]      = f2bf(p0[g]);
      pw[(hl * 8 + g) * VSP + 16 + cl] = f2bf(p1[g]);
    }
    asm volatile("s_wait_dscnt 0x0" ::: "memory");  // same-wave DS RAW

    v16bf pa = load_frag(pw, VSP, 0, lane);
#pragma unroll
    for (int dc = 0; dc < 32; ++dc) {
      v16bf vb = load_frag(VsB + dc * 16 * VSP, VSP, 0, lane);
      acc[dc] = __builtin_amdgcn_wmma_f32_16x16x32_bf16(
          false, pa, false, vb, (short)0, acc[dc], false, false);
    }
    // next iteration's wait+barrier protects the buffers
  }

  // ---- normalize + store O (bf16, row-major) ----
#pragma unroll
  for (int g = 0; g < 8; ++g) l_run[g] = 1.0f / l_run[g];
  us16* orow = Ob + ((size_t)n * S_LEN + qblk + wv * 16 + hl * 8) * E;
#pragma unroll
  for (int dc = 0; dc < 32; ++dc)
#pragma unroll
    for (int g = 0; g < 8; ++g)
      orow[(size_t)g * E + dc * 16 + cl] = f2bf(acc[dc][g] * l_run[g]);
}

// ---------------------------------------------------------------------------
extern "C" void kernel_launch(void* const* d_in, const int* in_sizes, int n_in,
                              void* d_out, int out_size, void* d_ws,
                              size_t ws_size, hipStream_t stream) {
  const float* values = (const float*)d_in[0];
  const float* keys   = (const float*)d_in[1];
  const float* query  = (const float*)d_in[2];
  const int*   mask   = (const int*)d_in[3];
  const float* Wv = (const float*)d_in[4];
  const float* Wk = (const float*)d_in[5];
  const float* Wq = (const float*)d_in[6];
  const float* Wo = (const float*)d_in[7];
  const float* bo = (const float*)d_in[8];
  float* out = (float*)d_out;

  us16* ws  = (us16*)d_ws;
  us16* Wvb = ws;
  us16* Wkb = Wvb + (size_t)E * E;
  us16* Wqb = Wkb + (size_t)E * E;
  us16* Wob = Wqb + (size_t)E * E;
  us16* Xq  = Wob + (size_t)E * E;
  us16* Xk  = Xq + (size_t)R * E;
  us16* Xv  = Xk + (size_t)R * E;
  us16* Qb  = Xv + (size_t)R * E;
  us16* Kb  = Qb + (size_t)R * E;
  us16* Vtb = Kb + (size_t)R * E;
  us16* Ob  = Vtb + (size_t)R * E;
  u32*  Mb  = (u32*)(Ob + (size_t)R * E);     // R*MW words (8.4 MB)

  const int WBLK = (E * E) / (256 * 4);       // 256 blocks per weight
  cvt_kernel<<<WBLK, 256, 0, stream>>>(Wv, Wvb);
  cvt_kernel<<<WBLK, 256, 0, stream>>>(Wk, Wkb);
  cvt_kernel<<<WBLK, 256, 0, stream>>>(Wq, Wqb);
  cvt_kernel<<<WBLK, 256, 0, stream>>>(Wo, Wob);

  const int XBLK = ((size_t)R * E) / (256 * 4);  // 8192 blocks per activation
  cvt_kernel<<<XBLK, 256, 0, stream>>>(query,  Xq);
  cvt_kernel<<<XBLK, 256, 0, stream>>>(keys,   Xk);
  cvt_kernel<<<XBLK, 256, 0, stream>>>(values, Xv);

  // 2M words / (8 per wave * 8 waves per block)
  mask_pack_kernel<<<((size_t)R * S_LEN / 32) / 64, 256, 0, stream>>>(mask, Mb);

  dim3 ggrid(R / 128, E / 64);
  gemm512_kernel<<<ggrid, 256, 0, stream>>>(Xq, Wqb, nullptr, Qb,  0);
  gemm512_kernel<<<ggrid, 256, 0, stream>>>(Xk, Wkb, nullptr, Kb,  0);
  gemm512_kernel<<<ggrid, 256, 0, stream>>>(Xv, Wvb, nullptr, Vtb, 1);

  flash_kernel<<<dim3(S_LEN / 128, N_B), 256, 0, stream>>>(Qb, Kb, Vtb, Mb, Ob);

  gemm512_kernel<<<ggrid, 256, 0, stream>>>(Ob, Wob, bo, out, 2);
}
